// MResConv_42949673553
// MI455X (gfx1250) — matrix operand order repaired and checked by reading, not compiled, not run on hardware
//
#include <hip/hip_runtime.h>

typedef __attribute__((ext_vector_type(4)))  float  v4f;
typedef __attribute__((ext_vector_type(8)))  float  v8f;
typedef __attribute__((ext_vector_type(8)))  __bf16 v8bf;
typedef __attribute__((ext_vector_type(16))) __bf16 v16bf;

#define MTILE 128    // block M tile (Cout slice)
#define ETILE 32     // block N tile (edges)
#define CCH   32     // channels per K-chunk
#define KCH   160    // K per chunk = CCH*5 (5 WMMA K=32 steps)
#define LSTR  168    // LDS row stride in bf16 (pad: 84 dwords, 16B aligned, few bank conflicts)

// A fragment (16x32 bf16): lane m=row; halves at K = ks+8*hi and ks+8*hi+16.
static __device__ inline v16bf make_frag(const __bf16* p) {
    union { v16bf v; v8bf h[2]; } u;
    u.h[0] = *(const v8bf*)(p);
    u.h[1] = *(const v8bf*)(p + 16);
    return u.v;
}
// B fragment (32x16 bf16): lane n=row; halves at K = ks+8*hi and ks+8*hi+16.
static __device__ inline v16bf make_bfrag(const __bf16* p) {
    union { v16bf v; v8bf h[2]; } u;
    u.h[0] = *(const v8bf*)(p);
    u.h[1] = *(const v8bf*)(p + 16);
    return u.v;
}

// Split fp32 weights into bf16 hi/lo pair: w = hi + lo + O(2^-17)
__global__ __launch_bounds__(256)
void split_w(const float* __restrict__ w, __bf16* __restrict__ hi,
             __bf16* __restrict__ lo, int n)
{
    int i = blockIdx.x * 256 + threadIdx.x;
    if (i < n) {
        float x = w[i];
        __bf16 h = (__bf16)x;
        hi[i] = h;
        lo[i] = (__bf16)(x - (float)h);
    }
}

// GEMM tile kernel for both mesh_conv layers, fp32-via-3xBF16 WMMA.
// X: features [B][CIN][E]; Whi/Wlo: [256][CIN*5] bf16 split weights.
// SECOND: apply relu+BN to gathered features; epilogue relu(acc + Y0res) -> OUT.
template<int CIN, bool SECOND>
__global__ __launch_bounds__(256)
void meshconv_gemm(const float*  __restrict__ X,
                   const int*    __restrict__ gemmIdx,   // [B][E][4]
                   const __bf16* __restrict__ Whi,
                   const __bf16* __restrict__ Wlo,
                   const float*  __restrict__ scale,
                   const float*  __restrict__ bias,
                   const float*  __restrict__ Y0res,
                   float* __restrict__ OUT,
                   int E)
{
    __shared__ __bf16 Wh[MTILE * LSTR];
    __shared__ __bf16 Wl[MTILE * LSTR];
    __shared__ __bf16 Gh[ETILE * LSTR];
    __shared__ __bf16 Gl[ETILE * LSTR];
    __shared__ int    idxs[ETILE * 4];

    const int tid  = threadIdx.x;
    const int lane = tid & 31;
    const int wid  = tid >> 5;
    const int b    = blockIdx.z;
    const int mblk = blockIdx.y * MTILE;
    const int nblk = blockIdx.x * ETILE;

    if (tid < ETILE * 4) {
        int e = tid >> 2, j = tid & 3;
        idxs[tid] = gemmIdx[((size_t)b * E + (nblk + e)) * 4 + j];
    }

    const int mw  = (wid >> 1) * 32;  // 4 waves along M, 2 along N
    const int nw  = (wid & 1) * 16;
    const int row = lane & 15;
    const int hi  = lane >> 4;

    v8f acc0 = {0.f,0.f,0.f,0.f,0.f,0.f,0.f,0.f};
    v8f acc1 = acc0;

    const int KTOT = CIN * 5;
    for (int c0 = 0; c0 < CIN; c0 += CCH) {
        __syncthreads();
        // ---- stage W chunk (bf16 hi/lo): rows [mblk,mblk+128), K in [c0*5, c0*5+160)
        {
            int r = tid >> 1;                      // 0..127
            int h = tid & 1;                       // half-row: 80 bf16 each
            size_t srcOff = (size_t)(mblk + r) * KTOT + c0 * 5 + h * 80;
            int    dstOff = r * LSTR + h * 80;
            #pragma unroll
            for (int q = 0; q < 10; ++q) {        // 10 x 16B per array
                *(v8bf*)(&Wh[dstOff + q * 8]) = *(const v8bf*)(Whi + srcOff + q * 8);
                *(v8bf*)(&Wl[dstOff + q * 8]) = *(const v8bf*)(Wlo + srcOff + q * 8);
            }
        }
        // ---- build G chunk: gather + symmetric fns + bf16 split; 32 ch x 32 edges
        {
            int cl = tid >> 5;                    // 0..7; covers 4 channels each
            int el = tid & 31;
            #pragma unroll
            for (int i = 0; i < 4; ++i) {
                int crel = cl + 8 * i;            // channel within chunk 0..31
                int c    = c0 + crel;
                const float* base = X + ((size_t)b * CIN + c) * E;
                float f0 = base[nblk + el];
                float f1 = base[idxs[el * 4 + 0]];
                float f2 = base[idxs[el * 4 + 1]];
                float f3 = base[idxs[el * 4 + 2]];
                float f4 = base[idxs[el * 4 + 3]];
                if (SECOND) {                     // fused relu -> batchnorm
                    float sc = scale[c], bi = bias[c];
                    f0 = fmaxf(f0, 0.f) * sc + bi;
                    f1 = fmaxf(f1, 0.f) * sc + bi;
                    f2 = fmaxf(f2, 0.f) * sc + bi;
                    f3 = fmaxf(f3, 0.f) * sc + bi;
                    f4 = fmaxf(f4, 0.f) * sc + bi;
                }
                float g[5];
                g[0] = f0;
                g[1] = f1 + f3;
                g[2] = f2 + f4;
                g[3] = fabsf(f1 - f3);
                g[4] = fabsf(f2 - f4);
                int off = el * LSTR + crel * 5;
                #pragma unroll
                for (int s = 0; s < 5; ++s) {
                    __bf16 hh = (__bf16)g[s];
                    Gh[off + s] = hh;
                    Gl[off + s] = (__bf16)(g[s] - (float)hh);
                }
            }
        }
        __syncthreads();
        // ---- 3xBF16 WMMA over this K chunk: acc += Ah*Bh + Ah*Bl + Al*Bh
        #pragma unroll
        for (int ks = 0; ks < KCH; ks += 32) {
            const __bf16* gp = &Gh[(nw + row) * LSTR + ks + 8 * hi];
            const __bf16* glp = &Gl[(nw + row) * LSTR + ks + 8 * hi];
            v16bf bh = make_bfrag(gp);
            v16bf bl = make_bfrag(glp);
            int aoff = (mw + row) * LSTR + ks + 8 * hi;
            v16bf ah0 = make_frag(&Wh[aoff]);
            v16bf al0 = make_frag(&Wl[aoff]);
            v16bf ah1 = make_frag(&Wh[aoff + 16 * LSTR]);
            v16bf al1 = make_frag(&Wl[aoff + 16 * LSTR]);
            acc0 = __builtin_amdgcn_wmma_f32_16x16x32_bf16(false, ah0, false, bh, (short)0, acc0, false, false);
            acc1 = __builtin_amdgcn_wmma_f32_16x16x32_bf16(false, ah1, false, bh, (short)0, acc1, false, false);
            acc0 = __builtin_amdgcn_wmma_f32_16x16x32_bf16(false, ah0, false, bl, (short)0, acc0, false, false);
            acc1 = __builtin_amdgcn_wmma_f32_16x16x32_bf16(false, ah1, false, bl, (short)0, acc1, false, false);
            acc0 = __builtin_amdgcn_wmma_f32_16x16x32_bf16(false, al0, false, bh, (short)0, acc0, false, false);
            acc1 = __builtin_amdgcn_wmma_f32_16x16x32_bf16(false, al1, false, bh, (short)0, acc1, false, false);
        }
    }

    // ---- epilogue: C layout -> VGPR r: M = r + 8*hi, N = row
    int n    = nblk + nw + row;
    int msub = hi * 8;
    #pragma unroll
    for (int r = 0; r < 8; ++r) {
        int m0 = mblk + mw + msub + r;
        int m1 = m0 + 16;
        size_t o0 = ((size_t)b * 256 + m0) * E + n;
        size_t o1 = ((size_t)b * 256 + m1) * E + n;
        if (SECOND) {
            OUT[o0] = fmaxf(acc0[r] + Y0res[o0], 0.f);
            OUT[o1] = fmaxf(acc1[r] + Y0res[o1], 0.f);
        } else {
            OUT[o0] = acc0[r];
            OUT[o1] = acc1[r];
        }
    }
}

// Per-channel BN(train) stats over relu(Y0), folded into scale/bias.
__global__ __launch_bounds__(256)
void bn_stats(const float* __restrict__ Y0, const float* __restrict__ gamma,
              const float* __restrict__ beta, float* __restrict__ scale,
              float* __restrict__ bias, int E, int B)
{
    __shared__ float ssum[256];
    __shared__ float ssq[256];
    const int c = blockIdx.x;
    float s = 0.f, q = 0.f;
    const int N = B * E;
    for (int i = threadIdx.x; i < N; i += 256) {
        int bb = i / E, e = i - bb * E;
        float v = Y0[((size_t)bb * 256 + c) * E + e];
        v = fmaxf(v, 0.f);
        s += v; q += v * v;
    }
    ssum[threadIdx.x] = s; ssq[threadIdx.x] = q;
    __syncthreads();
    for (int st = 128; st > 0; st >>= 1) {
        if (threadIdx.x < st) {
            ssum[threadIdx.x] += ssum[threadIdx.x + st];
            ssq[threadIdx.x]  += ssq[threadIdx.x + st];
        }
        __syncthreads();
    }
    if (threadIdx.x == 0) {
        float inv  = 1.f / (float)N;
        float mean = ssum[0] * inv;
        float var  = ssq[0] * inv - mean * mean;
        float sc   = gamma[c] * rsqrtf(var + 1e-5f);
        scale[c]   = sc;
        bias[c]    = beta[c] - mean * sc;
    }
}

extern "C" void kernel_launch(void* const* d_in, const int* in_sizes, int n_in,
                              void* d_out, int out_size, void* d_ws, size_t ws_size,
                              hipStream_t stream) {
    const float* x     = (const float*)d_in[0];   // [4][128][E]
    const int*   gemm  = (const int*)  d_in[1];   // [4][E][4]
    const float* w0    = (const float*)d_in[2];   // [256][128][5]
    const float* w1    = (const float*)d_in[3];   // [256][256][5]
    const float* gamma = (const float*)d_in[4];   // [256]
    const float* beta  = (const float*)d_in[5];   // [256]
    float*       out   = (float*)d_out;           // [4][256][E]

    const int B = 4, Cin = 128, Cout = 256;
    const int E = in_sizes[0] / (B * Cin);        // 20000
    const int n0 = Cout * Cin * 5;                // 163840
    const int n1 = Cout * Cout * 5;               // 327680

    // workspace layout (all 16B aligned)
    float*  Y0    = (float*)d_ws;                 // [4][256][E] pre-relu conv0 out
    float*  scale = Y0 + (size_t)B * Cout * E;
    float*  bias  = scale + Cout;
    __bf16* w0h   = (__bf16*)(bias + Cout);
    __bf16* w0l   = w0h + n0;
    __bf16* w1h   = w0l + n0;
    __bf16* w1l   = w1h + n1;

    dim3 blk(256);
    split_w<<<dim3(n0 / 256), blk, 0, stream>>>(w0, w0h, w0l, n0);
    split_w<<<dim3(n1 / 256), blk, 0, stream>>>(w1, w1h, w1l, n1);

    dim3 grd(E / ETILE, Cout / MTILE, B);         // 625 x 2 x 4
    meshconv_gemm<128, false><<<grd, blk, 0, stream>>>(x, gemm, w0h, w0l, nullptr, nullptr, nullptr, Y0, E);
    bn_stats<<<dim3(Cout), blk, 0, stream>>>(Y0, gamma, beta, scale, bias, E, B);
    meshconv_gemm<256, true><<<grd, blk, 0, stream>>>(Y0, gemm, w1h, w1l, scale, bias, Y0, out, E);
}
